// MultiHeadAttentionLayer_45990509806272
// MI455X (gfx1250) — matrix operand order repaired and checked by reading, not compiled, not run on hardware
//
#include <hip/hip_runtime.h>

#define HID 768
#define NH 12
#define HD 64
#define BB 2
#define SS 2048
#define SPAD 2064   // 2048 + 16 pad, keeps 16B row alignment in LDS

typedef __attribute__((ext_vector_type(16))) _Float16 v16h;
typedef __attribute__((ext_vector_type(8)))  _Float16 v8h;
typedef __attribute__((ext_vector_type(8)))  float    v8f;

__device__ __forceinline__ v8f wmma_f16(v16h a, v16h b, v8f c) {
  // (neg_a, A, neg_b, B, c_mod, C, reuse_a, reuse_b)
  return __builtin_amdgcn_wmma_f32_16x16x32_f16(false, a, false, b, (short)0, c,
                                                false, false);
}

// Raw 32-bit LDS byte offset of a generic pointer into __shared__ memory.
typedef __attribute__((address_space(3))) char lds_char;
__device__ __forceinline__ unsigned lds_off(const void* p) {
  return (unsigned)(unsigned long long)(lds_char*)p;
}

// ---------------------------------------------------------------------------
// Kernel 1: fused Q/K/V projection  Y = X * W^T + b  (f16 WMMA, fp32 accum)
// Q,K -> [B,NH,S,64] f16 ; V -> transposed [B,NH,64,S] f16
// one wave per 16x16 output tile; 3*256*48 = 36864 tiles; 9216 blocks x 128 thr
// ---------------------------------------------------------------------------
__global__ __launch_bounds__(128) void proj_kernel(
    const float* __restrict__ Xq, const float* __restrict__ Xk,
    const float* __restrict__ Xv, const float* __restrict__ W,
    const float* __restrict__ bias,
    _Float16* __restrict__ Qh, _Float16* __restrict__ Kh,
    _Float16* __restrict__ Vt)
{
  const int wave = threadIdx.x >> 5;
  const int lane = threadIdx.x & 31;
  const int m    = lane & 15;
  const int half = lane >> 4;

  const int tile  = blockIdx.x * 4 + wave;
  const int which = tile / (256 * 48);
  const int rem   = tile % (256 * 48);
  const int rt    = rem / 48;   // 16-row tile of [4096]
  const int ct    = rem % 48;   // 16-col tile of [768]

  const float* X = (which == 0) ? Xq : ((which == 1) ? Xk : Xv);
  const int row    = rt * 16 + m;   // A-fragment row for this lane
  const int outcol = ct * 16 + m;   // B/C column for this lane

  v8f c = {};
  for (int kk = 0; kk < 24; ++kk) {
    const int k0 = kk * 32;
    const float* xr = X + (size_t)row * HID + k0;
    v16h a, b;
#pragma unroll
    for (int j = 0; j < 8; ++j) {
      a[j]     = (_Float16)xr[16 * half + j];
      a[8 + j] = (_Float16)xr[16 * half + 8 + j];
    }
    const float* wr = W + (size_t)outcol * HID + k0 + 16 * half;
#pragma unroll
    for (int e = 0; e < 16; ++e) b[e] = (_Float16)wr[e];
    c = wmma_f16(a, b, c);
  }

  const float bv = bias[outcol];
#pragma unroll
  for (int r = 0; r < 8; ++r) c[r] += bv;

  const int hh = outcol / HD, d = outcol % HD;
  const int b2 = (rt * 16) / SS;          // whole tile is in one batch
  const int sbase = rt * 16 - b2 * SS;

  if (which == 2) {  // V transposed: [B,NH,64,S]; 8 contiguous f16 -> one 16B store
    _Float16* dst = Vt + (((size_t)b2 * NH + hh) * HD + d) * SS;
    v8h pk;
#pragma unroll
    for (int r = 0; r < 8; ++r) pk[r] = (_Float16)c[r];
    *(v8h*)(dst + sbase + 8 * half) = pk;
  } else {           // Q/K: [B,NH,S,64]
    _Float16* dst = (which == 0 ? Qh : Kh) +
                    ((size_t)b2 * NH + hh) * SS * HD + (size_t)d;
#pragma unroll
    for (int r = 0; r < 8; ++r)
      dst[(size_t)(sbase + 8 * half + r) * HD] = (_Float16)c[r];
  }
}

// ---------------------------------------------------------------------------
// Kernel 2: attention. One block (4 waves) per (b, h, 16-query strip).
// Scores 16x2048 in LDS (fp32), full softmax, attn streamed out via
// GLOBAL_STORE_ASYNC_FROM_LDS_B128 (no VGPR round trip, NT hint),
// then P*V with WMMA -> merged-head f16 output [4096,768].
// ---------------------------------------------------------------------------
__global__ __launch_bounds__(128) void attn_kernel(
    const _Float16* __restrict__ Qh, const _Float16* __restrict__ Kh,
    const _Float16* __restrict__ Vt, const int* __restrict__ mask,
    float* __restrict__ attn, _Float16* __restrict__ Of)
{
  __shared__ __align__(16) float sc[16][SPAD];
  __shared__ float red[16][8];
  __shared__ float rowv[16];

  const int tid  = threadIdx.x;
  const int wave = tid >> 5, lane = tid & 31;
  const int m    = lane & 15, half = lane >> 4;

  const int qt = blockIdx.x & 127;
  const int hh = (blockIdx.x >> 7) % NH;
  const int b  = blockIdx.x / (128 * NH);
  const size_t bh = (size_t)b * NH + hh;

  // Q fragments (16 queries x 64d) held in registers
  const _Float16* qr = Qh + (bh * SS + qt * 16 + m) * HD;
  v16h a0, a1;
#pragma unroll
  for (int j = 0; j < 8; ++j) {
    a0[j]     = qr[16 * half + j];
    a0[8 + j] = qr[16 * half + 8 + j];
    a1[j]     = qr[32 + 16 * half + j];
    a1[8 + j] = qr[40 + 16 * half + j];
  }

  // ---- scores: each wave owns 512 key columns (32 tiles of 16) ----
  for (int i = 0; i < 32; ++i) {
    const int nt  = wave * 32 + i;
    const int key = nt * 16 + m;
    const _Float16* kr = Kh + (bh * SS + key) * HD;
    v16h b0 = *(const v16h*)(kr + 16 * half);        // d = 0..31 slice
    v16h b1 = *(const v16h*)(kr + 32 + 16 * half);   // d = 32..63 slice
    v8f c = {};
    c = wmma_f16(a0, b0, c);
    c = wmma_f16(a1, b1, c);
    const int mv = mask[b * SS + key];               // same key for all 8 rows
#pragma unroll
    for (int r = 0; r < 8; ++r) {
      float e = c[r] * 0.125f;                       // 1/sqrt(64)
      sc[r + 8 * half][nt * 16 + m] = (mv == 0) ? -10000.0f : e;
    }
  }
  __syncthreads();

  // ---- softmax (fp32, in LDS): 8 threads per row, 256 cols each ----
  const int srow = tid >> 3, seg = tid & 7, c0 = seg * 256;
  float mx = -3.4e38f;
  for (int cc = 0; cc < 256; ++cc) mx = fmaxf(mx, sc[srow][c0 + cc]);
  red[srow][seg] = mx;
  __syncthreads();
  if (seg == 0) {
    float mm = red[srow][0];
#pragma unroll
    for (int j = 1; j < 8; ++j) mm = fmaxf(mm, red[srow][j]);
    rowv[srow] = mm;
  }
  __syncthreads();
  const float rm = rowv[srow];
  float sum = 0.f;
  for (int cc = 0; cc < 256; ++cc) {
    float e = __expf(sc[srow][c0 + cc] - rm);
    sc[srow][c0 + cc] = e;
    sum += e;
  }
  red[srow][seg] = sum;
  __syncthreads();
  if (seg == 0) {
    float s2 = 0.f;
#pragma unroll
    for (int j = 0; j < 8; ++j) s2 += red[srow][j];
    rowv[srow] = 1.0f / s2;
  }
  __syncthreads();
  const float inv = rowv[srow];
  for (int cc = 0; cc < 256; ++cc) sc[srow][c0 + cc] *= inv;
  __syncthreads();

  // ---- stream attn strip LDS -> HBM via async DMA (512 B/instr/wave) ----
  // Each wave owns 4 rows; per row: 8192 B = 16 instructions of 32 lanes x 16 B.
  {
    const unsigned scbase = lds_off(&sc[0][0]);
#pragma unroll
    for (int r0 = 0; r0 < 4; ++r0) {
      const int r = wave * 4 + r0;
      unsigned laddr = scbase + (unsigned)r * (SPAD * 4u) + (unsigned)lane * 16u;
      unsigned long long gaddr =
          (unsigned long long)(attn + (bh * SS + (size_t)(qt * 16 + r)) * SS) +
          (unsigned long long)lane * 16ull;
      for (int it = 0; it < 16; ++it) {
        asm volatile(
            "global_store_async_from_lds_b128 %0, %1, off th:TH_STORE_NT"
            :: "v"(gaddr + (unsigned long long)it * 512ull),
               "v"(laddr + (unsigned)it * 512u)
            : "memory");
      }
    }
  }

  // ---- O = P * V : each wave owns a 16-wide d-tile, K-loop over 2048 ----
  const int dcol = wave * 16 + m;
  const _Float16* vr = Vt + (bh * HD + dcol) * SS;
  v8f o = {};
  for (int kk = 0; kk < 64; ++kk) {
    const int k0 = kk * 32;
    v16h a, bf;
#pragma unroll
    for (int j = 0; j < 8; ++j) {
      a[j]     = (_Float16)sc[m][k0 + 16 * half + j];
      a[8 + j] = (_Float16)sc[m][k0 + 16 * half + 8 + j];
    }
    bf = *(const v16h*)(vr + k0 + 16 * half);
    o = wmma_f16(a, bf, o);
  }
  const int outc = hh * HD + dcol;
#pragma unroll
  for (int r = 0; r < 8; ++r) {
    const int qrow = qt * 16 + r + 8 * half;
    Of[((size_t)b * SS + qrow) * HID + outc] = (_Float16)o[r];
  }

  // drain async LDS->global stores before the wave can retire / LDS reuse
  asm volatile("s_wait_asynccnt 0" ::: "memory");
}

// ---------------------------------------------------------------------------
// Kernel 3: output projection  out = O * W^T + b  -> fp32 d_out
// ---------------------------------------------------------------------------
__global__ __launch_bounds__(128) void outproj_kernel(
    const _Float16* __restrict__ Of, const float* __restrict__ W,
    const float* __restrict__ bias, float* __restrict__ out)
{
  const int wave = threadIdx.x >> 5, lane = threadIdx.x & 31;
  const int m = lane & 15, half = lane >> 4;
  const int tile = blockIdx.x * 4 + wave;
  const int rt = tile / 48, ct = tile % 48;
  const int row = rt * 16 + m;
  const int outcol = ct * 16 + m;

  v8f c = {};
  for (int kk = 0; kk < 24; ++kk) {
    const int k0 = kk * 32;
    const _Float16* xr = Of + (size_t)row * HID + k0;
    v16h a, b;
#pragma unroll
    for (int j = 0; j < 8; ++j) {
      a[j]     = xr[16 * half + j];
      a[8 + j] = xr[16 * half + 8 + j];
    }
    const float* wr = W + (size_t)outcol * HID + k0 + 16 * half;
#pragma unroll
    for (int e = 0; e < 16; ++e) b[e] = (_Float16)wr[e];
    c = wmma_f16(a, b, c);
  }
  const float bv = bias[outcol];
#pragma unroll
  for (int r = 0; r < 8; ++r)
    out[(size_t)(rt * 16 + r + 8 * half) * HID + outcol] = c[r] + bv;
}

// ---------------------------------------------------------------------------
extern "C" void kernel_launch(void* const* d_in, const int* in_sizes, int n_in,
                              void* d_out, int out_size, void* d_ws, size_t ws_size,
                              hipStream_t stream)
{
  const float* q    = (const float*)d_in[0];
  const float* k    = (const float*)d_in[1];
  const float* v    = (const float*)d_in[2];
  const int*   mask = (const int*)d_in[3];
  const float* W    = (const float*)d_in[4];
  const float* bias = (const float*)d_in[5];

  float* out  = (float*)d_out;                     // [B,S,HID] = 3,145,728 f32
  float* attn = out + (size_t)BB * SS * HID;       // [B,NH,S,S]

  const size_t nQ = (size_t)BB * NH * SS * HD;     // 3,145,728 f16 elements
  _Float16* Qh = (_Float16*)d_ws;
  _Float16* Kh = Qh + nQ;
  _Float16* Vt = Kh + nQ;
  _Float16* Of = Vt + nQ;                          // total ws use ~25.2 MB

  proj_kernel<<<dim3(9216), dim3(128), 0, stream>>>(q, k, v, W, bias, Qh, Kh, Vt);
  attn_kernel<<<dim3(3072), dim3(128), 0, stream>>>(Qh, Kh, Vt, mask, attn, Of);
  outproj_kernel<<<dim3(3072), dim3(128), 0, stream>>>(Of, W, bias, out);

  (void)in_sizes; (void)n_in; (void)out_size; (void)ws_size;
}
